// W2v_Attention_21388937134133
// MI455X (gfx1250) — matrix operand order repaired and checked by reading, not compiled, not run on hardware
//
#include <hip/hip_runtime.h>

// ---------------------------------------------------------------------------
// W2v attention block for MI455X (gfx1250, wave32, WMMA)
//   qn = LN(q) @ Wqkv^T ; kn = LN(k) @ Wqkv^T ; vn = LN(v) @ Wqkv^T
//   out = softmax(qn kn^T / sqrt(D)) vn @ Wproj^T + b
// N = 8192, D = 768.  All GEMMs via v_wmma_f32_16x16x32_f16.
// ---------------------------------------------------------------------------

#define NN_ROWS 8192
#define DD      768
#define SCALE   0.03608439182435161f   // 1/sqrt(768)

typedef __attribute__((ext_vector_type(16))) _Float16 v16h;
typedef __attribute__((ext_vector_type(8)))  _Float16 v8h;
typedef __attribute__((ext_vector_type(8)))  float    v8f;
typedef __attribute__((ext_vector_type(4)))  int      v4i;

#if __has_builtin(__builtin_amdgcn_global_load_async_to_lds_b128)
#define HAVE_ASYNC_LDS 1
#else
#define HAVE_ASYNC_LDS 0
#endif

#define AS_GLOBAL __attribute__((address_space(1)))
#define AS_SHARED __attribute__((address_space(3)))

union AFragU { v16h v; v8h h[2]; };

// A-fragment (16x32 f16, M x K): lane L holds row M=(L&15);
// halves 0..15 = K = kb .. kb+15,
// where kb = k0 + (L>=16 ? 16 : 0).   (matches B-fragment K order)
__device__ __forceinline__ v16h load_a_frag(const _Float16* __restrict__ A,
                                            int ld, int m0, int k0, int lane) {
  int m  = m0 + (lane & 15);
  int kb = k0 + ((lane >> 4) << 4);
  const _Float16* p = A + (size_t)m * ld + kb;
  AFragU f;
  f.h[0] = *(const v8h*)(p);
  f.h[1] = *(const v8h*)(p + 8);
  return f.v;
}

// B-fragment (32x16 f16, K x N) from a row-major [N,K] matrix (i.e. B = Bt^T):
// lane L holds col N=(L&15), 16 contiguous K values starting at k0+(L>=16?16:0).
__device__ __forceinline__ v16h load_b_frag(const _Float16* __restrict__ Bt,
                                            int ld, int n0, int k0, int lane) {
  int n  = n0 + (lane & 15);
  int kb = k0 + ((lane >> 4) << 4);
  return *(const v16h*)(Bt + (size_t)n * ld + kb);
}

__device__ __forceinline__ v8f wmma_f16(v16h a, v16h b, v8f c) {
  return __builtin_amdgcn_wmma_f32_16x16x32_f16(
      /*neg_a=*/false, a, /*neg_b=*/false, b,
      /*c_mod=*/(short)0, c, /*reuse_a=*/false, /*reuse_b=*/false);
}

// ---------------------------------------------------------------------------
// Kernel 1: LayerNorm (f32 in) -> f16 out.  One wave32 per row, 8 rows/block.
// ---------------------------------------------------------------------------
__global__ void ln_f16_kernel(const float* __restrict__ x,
                              const float* __restrict__ w,
                              const float* __restrict__ b,
                              _Float16* __restrict__ y) {
  int row  = blockIdx.x * 8 + (threadIdx.x >> 5);
  int lane = threadIdx.x & 31;
  const float* xr = x + (size_t)row * DD;
  float vals[24];
  float s = 0.f, s2 = 0.f;
#pragma unroll
  for (int i = 0; i < 24; ++i) {
    float t = xr[lane + i * 32];
    vals[i] = t; s += t; s2 += t * t;
  }
#pragma unroll
  for (int off = 16; off > 0; off >>= 1) {
    s  += __shfl_xor(s,  off, 32);
    s2 += __shfl_xor(s2, off, 32);
  }
  float mu   = s * (1.f / DD);
  float var  = s2 * (1.f / DD) - mu * mu;
  float rstd = rsqrtf(var + 1e-5f);
  _Float16* yr = y + (size_t)row * DD;
#pragma unroll
  for (int i = 0; i < 24; ++i) {
    int c = lane + i * 32;
    yr[c] = (_Float16)((vals[i] - mu) * rstd * w[c] + b[c]);
  }
}

// ---------------------------------------------------------------------------
// Kernel 2: f32 -> f16 convert (weights).
// ---------------------------------------------------------------------------
__global__ void cvt_f16_kernel(const float* __restrict__ x,
                               _Float16* __restrict__ y, int n) {
  int i = blockIdx.x * 256 + threadIdx.x;
  if (i < n) y[i] = (_Float16)x[i];
}

// ---------------------------------------------------------------------------
// Kernel 3: C = A[M,K] @ Bt[N,K]^T, all WMMA.
// Block = 256 thr = 8 waves (4 M x 2 N), wave tile 32x32 -> block tile 128x64.
// Epilogues: f16 (optionally transposed, for vt) or f32+bias (projection).
// ---------------------------------------------------------------------------
__global__ void gemm_wmma_kernel(const _Float16* __restrict__ A,
                                 const _Float16* __restrict__ Bt,
                                 _Float16* __restrict__ C16,
                                 float* __restrict__ C32,
                                 const float* __restrict__ bias,
                                 int Nn, int K, int ldt, int transC) {
  int lane = threadIdx.x & 31;
  int wave = threadIdx.x >> 5;
  int m0 = blockIdx.x * 128 + (wave & 3) * 32;
  int n0 = blockIdx.y * 64  + (wave >> 2) * 32;

  v8f acc[2][2] = {};
  for (int k0 = 0; k0 < K; k0 += 32) {
    v16h a0 = load_a_frag(A, K, m0,      k0, lane);
    v16h a1 = load_a_frag(A, K, m0 + 16, k0, lane);
    v16h b0 = load_b_frag(Bt, K, n0,      k0, lane);
    v16h b1 = load_b_frag(Bt, K, n0 + 16, k0, lane);
    acc[0][0] = wmma_f16(a0, b0, acc[0][0]);
    acc[0][1] = wmma_f16(a0, b1, acc[0][1]);
    acc[1][0] = wmma_f16(a1, b0, acc[1][0]);
    acc[1][1] = wmma_f16(a1, b1, acc[1][1]);
  }
  // C/D layout: VGPR r -> (M = r + 8*(lane>=16), N = lane&15)
  int rb = (lane >> 4) << 3;
  int nc = lane & 15;
#pragma unroll
  for (int i = 0; i < 2; ++i)
#pragma unroll
    for (int j = 0; j < 2; ++j) {
      int nn = n0 + j * 16 + nc;
#pragma unroll
      for (int r = 0; r < 8; ++r) {
        int mm = m0 + i * 16 + rb + r;
        float val = acc[i][j][r];
        if (C32) {
          C32[(size_t)mm * Nn + nn] = val + (bias ? bias[nn] : 0.f);
        } else if (transC) {
          C16[(size_t)nn * ldt + mm] = (_Float16)val;
        } else {
          C16[(size_t)mm * Nn + nn] = (_Float16)val;
        }
      }
    }
}

// ---------------------------------------------------------------------------
// Kernel 4: attention for 16 query rows per block.
// Full logit row-block (16 x 8192 f16) lives in LDS (~256 KB of the 320 KB
// WGP LDS).  Phase 1: S = Q K^T (WMMA, Q staged in LDS via async-to-LDS when
// available).  Phase 2: exact softmax in-place.  Phase 3: O = P V (WMMA,
// A-frags from LDS, V transposed in memory so B-frags are contiguous 32B).
// ---------------------------------------------------------------------------
#define SROW 8200   // 8192 + 8 halves pad -> 16 A-frag rows hit distinct banks
#define QROW 776    // 768 + 8 halves pad

__global__ void attn_kernel(const _Float16* __restrict__ qn,
                            const _Float16* __restrict__ kn,
                            const _Float16* __restrict__ vt,   // [DD, NN_ROWS]
                            _Float16* __restrict__ o16) {
  extern __shared__ char smem[];
  _Float16* Sld = (_Float16*)smem;                 // 16 x SROW
  _Float16* Qld = Sld + 16 * SROW;                 // 16 x QROW
  float* red    = (float*)(Qld + 16 * QROW);       // 16 x 16 partials
  float* rowsum = red + 256;                       // 16 inverse row sums

  const int q0   = blockIdx.x * 16;
  const int tid  = threadIdx.x;
  const int lane = tid & 31;
  const int wave = tid >> 5;

  // stage Q tile (16 x 768 f16) into LDS, 16B chunks, async DMA when available
  for (int idx = tid; idx < 16 * 96; idx += 256) {
    int r = idx / 96, c = (idx % 96) * 8;
    const _Float16* g = qn + (size_t)(q0 + r) * DD + c;
    _Float16* l = Qld + r * QROW + c;
#if HAVE_ASYNC_LDS
    __builtin_amdgcn_global_load_async_to_lds_b128(
        (AS_GLOBAL v4i*)g, (AS_SHARED v4i*)l, 0, 0);
#else
    *(v8h*)l = *(const v8h*)g;
#endif
  }
#if HAVE_ASYNC_LDS
#if __has_builtin(__builtin_amdgcn_s_wait_asynccnt)
  __builtin_amdgcn_s_wait_asynccnt(0);
#else
  asm volatile("s_wait_asynccnt 0x0" ::: "memory");
#endif
#endif
  __syncthreads();

  // ---- Phase 1: logits -------------------------------------------------
  const int rb = (lane >> 4) << 3;
  const int nc = lane & 15;
  for (int jt = wave; jt < NN_ROWS / 16; jt += 8) {
    int n0 = jt * 16;
    // prefetch next tile of kn (this wave's next jt) into cache
    if (jt + 8 < NN_ROWS / 16)
      __builtin_prefetch(kn + (size_t)(n0 + 128 + nc) * DD, 0, 1);
    v8f acc = {};
    for (int k0 = 0; k0 < DD; k0 += 32) {
      v16h a = load_a_frag(Qld, QROW, 0, k0, lane);      // ds_load_b128 x2
      v16h b = load_b_frag(kn, DD, n0, k0, lane);        // global 32B
      acc = wmma_f16(a, b, acc);
    }
#pragma unroll
    for (int r = 0; r < 8; ++r)
      Sld[(size_t)(rb + r) * SROW + n0 + nc] = (_Float16)(acc[r] * SCALE);
  }
  __syncthreads();

  // ---- Phase 2: softmax (exact, per row) -------------------------------
  {
    int row = tid >> 4, sub = tid & 15;
    _Float16* sr = Sld + (size_t)row * SROW;
    float m = -3.0e38f;
    for (int c = sub; c < NN_ROWS; c += 16) m = fmaxf(m, (float)sr[c]);
    red[row * 16 + sub] = m;
    __syncthreads();
    float mrow = red[row * 16];
#pragma unroll
    for (int i = 1; i < 16; ++i) mrow = fmaxf(mrow, red[row * 16 + i]);
    __syncthreads();
    float s = 0.f;
    for (int c = sub; c < NN_ROWS; c += 16) {
      float e = __expf((float)sr[c] - mrow);
      sr[c] = (_Float16)e;
      s += e;
    }
    red[row * 16 + sub] = s;
    __syncthreads();
    if (sub == 0) {
      float t = 0.f;
#pragma unroll
      for (int i = 0; i < 16; ++i) t += red[row * 16 + i];
      rowsum[row] = 1.f / t;
    }
    __syncthreads();
  }

  // ---- Phase 3: O = P V ------------------------------------------------
  for (int dt = wave; dt < DD / 16; dt += 8) {
    int n0 = dt * 16;
    v8f acc = {};
    for (int k0 = 0; k0 < NN_ROWS; k0 += 32) {
      if ((k0 & 255) == 0)   // prefetch one 256-wide K chunk ahead
        __builtin_prefetch(vt + (size_t)(n0 + nc) * NN_ROWS + k0 + 256, 0, 1);
      v16h a = load_a_frag(Sld, SROW, 0, k0, lane);      // P from LDS
      v16h b = load_b_frag(vt, NN_ROWS, n0, k0, lane);   // contiguous 32B
      acc = wmma_f16(a, b, acc);
    }
#pragma unroll
    for (int r = 0; r < 8; ++r) {
      int mm = rb + r;
      o16[(size_t)(q0 + mm) * DD + n0 + nc] =
          (_Float16)(acc[r] * rowsum[mm]);
    }
  }
}

// ---------------------------------------------------------------------------
// Host launch
// ---------------------------------------------------------------------------
extern "C" void kernel_launch(void* const* d_in, const int* in_sizes, int n_in,
                              void* d_out, int out_size, void* d_ws, size_t ws_size,
                              hipStream_t stream) {
  (void)in_sizes; (void)n_in; (void)out_size; (void)ws_size;
  const float* q      = (const float*)d_in[0];
  const float* k      = (const float*)d_in[1];
  const float* v      = (const float*)d_in[2];
  const float* ln_w   = (const float*)d_in[3];
  const float* ln_b   = (const float*)d_in[4];
  const float* W_qkv  = (const float*)d_in[5];
  const float* W_proj = (const float*)d_in[6];
  const float* b_proj = (const float*)d_in[7];
  float* out = (float*)d_out;

  const size_t ND = (size_t)NN_ROWS * DD;
  _Float16* ws    = (_Float16*)d_ws;
  _Float16* lnbuf = ws;            // reused for ln(q), ln(k), ln(v)
  _Float16* qn    = ws + ND;
  _Float16* kn    = ws + 2 * ND;
  _Float16* vt    = ws + 3 * ND;   // [DD, NN_ROWS] (transposed vn)
  _Float16* o16   = ws + 4 * ND;
  _Float16* w16   = ws + 5 * ND;           // Wqkv  f16
  _Float16* wp16  = w16 + (size_t)DD * DD; // Wproj f16

  dim3 blk(256);
  dim3 gW((DD * DD + 255) / 256);
  cvt_f16_kernel<<<gW, blk, 0, stream>>>(W_qkv,  w16,  DD * DD);
  cvt_f16_kernel<<<gW, blk, 0, stream>>>(W_proj, wp16, DD * DD);

  dim3 gLN(NN_ROWS / 8);
  dim3 gGEMM(NN_ROWS / 128, DD / 64);

  // qn = LN(q) @ Wqkv^T
  ln_f16_kernel<<<gLN, blk, 0, stream>>>(q, ln_w, ln_b, lnbuf);
  gemm_wmma_kernel<<<gGEMM, blk, 0, stream>>>(lnbuf, w16, qn, nullptr, nullptr,
                                              DD, DD, 0, 0);
  // kn = LN(k) @ Wqkv^T
  ln_f16_kernel<<<gLN, blk, 0, stream>>>(k, ln_w, ln_b, lnbuf);
  gemm_wmma_kernel<<<gGEMM, blk, 0, stream>>>(lnbuf, w16, kn, nullptr, nullptr,
                                              DD, DD, 0, 0);
  // vt = (LN(v) @ Wqkv^T)^T   (transposed store so PV B-frags are contiguous)
  ln_f16_kernel<<<gLN, blk, 0, stream>>>(v, ln_w, ln_b, lnbuf);
  gemm_wmma_kernel<<<gGEMM, blk, 0, stream>>>(lnbuf, w16, vt, nullptr, nullptr,
                                              DD, DD, NN_ROWS, 1);

  // attention: 512 blocks x 16 query rows, ~288 KB LDS each
  size_t attn_lds = (size_t)16 * SROW * 2 + (size_t)16 * QROW * 2 + 272 * 4;
  attn_kernel<<<dim3(NN_ROWS / 16), blk, attn_lds, stream>>>(qn, kn, vt, o16);

  // out = O @ Wproj^T + b
  gemm_wmma_kernel<<<gGEMM, blk, 0, stream>>>(o16, wp16, nullptr, out, b_proj,
                                              DD, DD, 0, 0);
}
